// MoELayer_22101901705553
// MI455X (gfx1250) — matrix operand order repaired and checked by reading, not compiled, not run on hardware
//
#include <hip/hip_runtime.h>
#include <hip/hip_bf16.h>
#include <math.h>

typedef __attribute__((ext_vector_type(16))) __bf16 v16bf;
typedef __attribute__((ext_vector_type(8)))  __bf16 v8bf;
typedef __attribute__((ext_vector_type(8)))  float  v8f;

#define T_TOKENS 16384      // B*S = 8*2048
#define DMODEL   768
#define HDIM     768
#define NEXP     4
#define TM       128
#define TN       128
#define TK       32
#define KSTEPS   (DMODEL / TK)            // 24
#define ROWCAP   (T_TOKENS + NEXP * TM)   // 16896 padded rows max
#define MAXMT    (T_TOKENS / TM + NEXP)   // 132 fixed M-tile slots
#define NTILES_N (HDIM / TN)              // 6

// ---------------- workspace layout (bytes) ----------------
#define XG_BYTES   ((size_t)ROWCAP * DMODEL * 2)
#define HB_BYTES   ((size_t)ROWCAP * HDIM * 2)
#define WB_BYTES   ((size_t)NEXP * HDIM * DMODEL * 2)
#define XG_OFF     ((size_t)0)
#define HB_OFF     (XG_OFF + XG_BYTES)
#define W1B_OFF    (HB_OFF + HB_BYTES)
#define W2B_OFF    (W1B_OFF + WB_BYTES)
#define EID_OFF    (W2B_OFF + WB_BYTES)
#define PSEL_OFF   (EID_OFF + (size_t)T_TOKENS * 4)
#define RMAP_OFF   (PSEL_OFF + (size_t)T_TOKENS * 4)
#define META_OFF   (RMAP_OFF + (size_t)ROWCAP * 4)    // 12 words: counts[4], cursors[4], probsum[4]
#define PADOFF_OFF (META_OFF + 256)                   // 5 ints
#define TILES_OFF  (PADOFF_OFF + 256)                 // MAXMT int2

// ---------------- CDNA5 async global -> LDS (ASYNCcnt-tracked) ----------------
__device__ __forceinline__ void async_ld_b128(unsigned lds_addr, const void* gaddr) {
  asm volatile("global_load_async_to_lds_b128 %0, %1, off"
               :: "v"(lds_addr), "v"((unsigned long long)(uintptr_t)gaddr)
               : "memory");
}
__device__ __forceinline__ unsigned lds_off(const void* p) {
  // generic pointers to LDS carry the LDS byte offset in the low 32 bits
  return (unsigned)(uintptr_t)p;
}

// ---------------- clear / init ----------------
__global__ void clear_kernel(uint4* __restrict__ xg4, int nx4,
                             int* __restrict__ rowmap, int* __restrict__ meta) {
  int id = blockIdx.x * blockDim.x + threadIdx.x;
  int stride = gridDim.x * blockDim.x;
  uint4 z; z.x = 0; z.y = 0; z.z = 0; z.w = 0;
  for (int i = id; i < nx4; i += stride) xg4[i] = z;
  for (int i = id; i < ROWCAP; i += stride) rowmap[i] = -1;
  if (id < 12) meta[id] = 0;
}

// ---------------- gating: one wave32 per token ----------------
__global__ void gate_kernel(const float* __restrict__ x, const float* __restrict__ gw,
                            int* __restrict__ eid, float* __restrict__ psel,
                            int* __restrict__ counts, float* __restrict__ probsum) {
  int t = blockIdx.x * 8 + (threadIdx.x >> 5);
  int lane = threadIdx.x & 31;
  const float* xr = x + (size_t)t * DMODEL;
  float a0 = 0.f, a1 = 0.f, a2 = 0.f, a3 = 0.f;
  for (int i = lane; i < DMODEL; i += 32) {
    float xv = xr[i];
    a0 += xv * gw[0 * DMODEL + i];
    a1 += xv * gw[1 * DMODEL + i];
    a2 += xv * gw[2 * DMODEL + i];
    a3 += xv * gw[3 * DMODEL + i];
  }
#pragma unroll
  for (int m = 16; m > 0; m >>= 1) {
    a0 += __shfl_xor(a0, m, 32);
    a1 += __shfl_xor(a1, m, 32);
    a2 += __shfl_xor(a2, m, 32);
    a3 += __shfl_xor(a3, m, 32);
  }
  if (lane == 0) {
    float l[NEXP] = {a0, a1, a2, a3};
    int be = 0; float best = l[0];
#pragma unroll
    for (int e = 1; e < NEXP; ++e) if (l[e] > best) { best = l[e]; be = e; }
    float s = 0.f;
    float p[NEXP];
#pragma unroll
    for (int e = 0; e < NEXP; ++e) { p[e] = __expf(l[e] - best); s += p[e]; }
    float inv = 1.f / s;
#pragma unroll
    for (int e = 0; e < NEXP; ++e) atomicAdd(&probsum[e], p[e] * inv);
    atomicAdd(&counts[be], 1);
    eid[t] = be;
    psel[t] = p[be] * inv;   // = 1/s since p[be]==1
  }
}

// ---------------- setup: padded offsets, tile table, loss tail ----------------
__global__ void setup_kernel(const int* __restrict__ counts, const float* __restrict__ probsum,
                             int* __restrict__ padoff, int2* __restrict__ tiles,
                             float* __restrict__ out_tail) {
  if (threadIdx.x != 0 || blockIdx.x != 0) return;
  int po[NEXP + 1]; po[0] = 0;
  for (int e = 0; e < NEXP; ++e) {
    int c = counts[e];
    po[e + 1] = po[e] + ((c + TM - 1) / TM) * TM;
  }
  for (int e = 0; e <= NEXP; ++e) padoff[e] = po[e];
  int slot = 0;
  for (int e = 0; e < NEXP; ++e)
    for (int r = po[e]; r < po[e + 1]; r += TM) { tiles[slot].x = r; tiles[slot].y = e; ++slot; }
  for (; slot < MAXMT; ++slot) { tiles[slot].x = 0; tiles[slot].y = -1; }
  // balance loss: E * sum_e (probsum[e]/B) * (counts[e]/T)
  float bal = 0.f;
  for (int e = 0; e < NEXP; ++e)
    bal += (probsum[e] / 8.0f) * ((float)counts[e] / (float)T_TOKENS);
  out_tail[0] = bal * (float)NEXP;
  for (int e = 0; e < NEXP; ++e) out_tail[1 + e] = (float)counts[e];
}

// ---------------- scatter: compact tokens per expert, fp32 -> bf16 ----------------
__global__ void scatter_kernel(const float* __restrict__ x, const int* __restrict__ eid,
                               int* __restrict__ cursors, const int* __restrict__ padoff,
                               int* __restrict__ rowmap, __bf16* __restrict__ Xg) {
  int t = blockIdx.x * 8 + (threadIdx.x >> 5);
  int lane = threadIdx.x & 31;
  int row = 0;
  if (lane == 0) {
    int e = eid[t];
    int pos = atomicAdd(&cursors[e], 1);
    row = padoff[e] + pos;
    rowmap[row] = t;
  }
  row = __shfl(row, 0, 32);
  const float* xr = x + (size_t)t * DMODEL;
  __bf16* dst = Xg + (size_t)row * DMODEL;
  for (int i = lane; i < DMODEL; i += 32) dst[i] = (__bf16)xr[i];
}

// ---------------- weight conversion fp32 -> bf16 ----------------
__global__ void convw_kernel(const float* __restrict__ w1, const float* __restrict__ w2,
                             __bf16* __restrict__ W1b, __bf16* __restrict__ W2b) {
  int n = NEXP * HDIM * DMODEL;
  int stride = gridDim.x * blockDim.x;
  for (int i = blockIdx.x * blockDim.x + threadIdx.x; i < n; i += stride) {
    W1b[i] = (__bf16)w1[i];
    W2b[i] = (__bf16)w2[i];
  }
}

// ---------------- expert-grouped bf16 WMMA GEMM ----------------
// C[128x128] per block; 8 waves, each 32x64 (2x4 WMMA tiles), K-step 32.
// Double-buffered LDS filled via GLOBAL_LOAD_ASYNC_TO_LDS_B128 (ASYNCcnt),
// next tile's fetch overlaps current tile's WMMAs.
// A: [ROWCAP, 768] bf16 row-major (gathered tokens or H buffer)
// W: [E][N=768][K=768] bf16, output-row-major, K contiguous -> no transpose.
template <bool IS_FFN1>
__global__ __launch_bounds__(256)
void gemm_kernel(const __bf16* __restrict__ A, const __bf16* __restrict__ W,
                 const float* __restrict__ bias, const int2* __restrict__ tiles,
                 const int* __restrict__ rowmap, const float* __restrict__ psel,
                 __bf16* __restrict__ Hout, float* __restrict__ out) {
  __shared__ __align__(128) __bf16 As[2][TM][TK];
  __shared__ __align__(128) __bf16 Bs[2][TN][TK];

  int2 td = tiles[blockIdx.y];
  int expert = td.y;
  if (expert < 0) return;
  int rowbase = td.x;
  int n0 = blockIdx.x * TN;

  const __bf16* Wm = W + ((size_t)expert * 768 + n0) * 768;

  int tid    = threadIdx.x;
  int ldrow  = tid >> 1;              // 0..127
  int ldhalf = (tid & 1) * 16;        // element offset 0 or 16
  int lane   = tid & 31;
  int wid    = tid >> 5;
  int wm     = (wid & 3) * 32;        // wave M offset in tile
  int wn     = (wid >> 2) * 64;       // wave N offset in tile
  int fm     = lane & 15;
  int fka    = (lane >> 4) * 16;      // A frag K base (contiguous K, matches B)
  int fkb    = (lane >> 4) * 16;      // B frag K base

  const __bf16* gA = A  + (size_t)(rowbase + ldrow) * 768 + ldhalf;
  const __bf16* gB = Wm + (size_t)ldrow * 768 + ldhalf;
  unsigned lA0[2], lA1[2], lB0[2], lB1[2];
#pragma unroll
  for (int b = 0; b < 2; ++b) {
    lA0[b] = lds_off(&As[b][ldrow][ldhalf]);
    lA1[b] = lds_off(&As[b][ldrow][ldhalf + 8]);
    lB0[b] = lds_off(&Bs[b][ldrow][ldhalf]);
    lB1[b] = lds_off(&Bs[b][ldrow][ldhalf + 8]);
  }

  v8f acc[2][4];
#pragma unroll
  for (int i = 0; i < 2; ++i)
#pragma unroll
    for (int j = 0; j < 4; ++j) acc[i][j] = (v8f)(0.0f);

  // prologue: fetch K-step 0 into buffer 0 (4 async ops per wave)
  async_ld_b128(lA0[0], gA);
  async_ld_b128(lA1[0], gA + 8);
  async_ld_b128(lB0[0], gB);
  async_ld_b128(lB1[0], gB + 8);

  for (int kk = 0; kk < KSTEPS; ++kk) {
    int buf = kk & 1;
    if (kk + 1 < KSTEPS) {
      int nb = (kk + 1) & 1;
      int ko = (kk + 1) * TK;
      async_ld_b128(lA0[nb], gA + ko);
      async_ld_b128(lA1[nb], gA + ko + 8);
      async_ld_b128(lB0[nb], gB + ko);
      async_ld_b128(lB1[nb], gB + ko + 8);
      asm volatile("s_wait_asynccnt 0x4" ::: "memory");  // current tile's 4 ops done
    } else {
      asm volatile("s_wait_asynccnt 0x0" ::: "memory");
    }
    __syncthreads();  // all waves' async writes for `buf` visible

    union { v16bf v; v8bf h[2]; } af[2];
#pragma unroll
    for (int i = 0; i < 2; ++i) {
      af[i].h[0] = *(const v8bf*)&As[buf][wm + i * 16 + fm][fka];
      af[i].h[1] = *(const v8bf*)&As[buf][wm + i * 16 + fm][fka + 8];
    }
    v16bf bfr[4];
#pragma unroll
    for (int j = 0; j < 4; ++j)
      bfr[j] = *(const v16bf*)&Bs[buf][wn + j * 16 + fm][fkb];

#pragma unroll
    for (int i = 0; i < 2; ++i)
#pragma unroll
      for (int j = 0; j < 4; ++j)
        acc[i][j] = __builtin_amdgcn_wmma_f32_16x16x32_bf16(
            false, af[i].v, false, bfr[j], (short)0, acc[i][j], false, false);

    __syncthreads();  // protect `buf` before iteration kk+1 refills it
  }

  // Epilogue. C layout: VGPR v, lanes0-15 -> M=v, N=lane; lanes16-31 -> M=v+8, N=lane-16.
  int rsel = (lane >> 4) * 8;
  int csel = lane & 15;
#pragma unroll
  for (int i = 0; i < 2; ++i) {
#pragma unroll
    for (int v = 0; v < 8; ++v) {
      int rloc = wm + i * 16 + rsel + v;
      int r = rowbase + rloc;
      if (IS_FFN1) {
#pragma unroll
        for (int j = 0; j < 4; ++j) {
          int c = n0 + wn + j * 16 + csel;
          float val = acc[i][j][v] + bias[expert * HDIM + c];
          float g = 0.5f * val * (1.0f + erff(val * 0.70710678118f));
          Hout[(size_t)r * HDIM + c] = (__bf16)g;
        }
      } else {
        int t = rowmap[r];
        if (t >= 0) {
          float p = psel[t];
#pragma unroll
          for (int j = 0; j < 4; ++j) {
            int c = n0 + wn + j * 16 + csel;
            float val = acc[i][j][v] + bias[expert * DMODEL + c];
            out[(size_t)t * DMODEL + c] = val * p;
          }
        }
      }
    }
  }
}

// ---------------- host ----------------
extern "C" void kernel_launch(void* const* d_in, const int* in_sizes, int n_in,
                              void* d_out, int out_size, void* d_ws, size_t ws_size,
                              hipStream_t stream) {
  const float* x      = (const float*)d_in[0];
  // d_in[1] input_ids, d_in[2] attention_mask: unused by the math
  const float* gate_w = (const float*)d_in[3];
  const float* w1     = (const float*)d_in[4];
  const float* b1     = (const float*)d_in[5];
  const float* w2     = (const float*)d_in[6];
  const float* b2     = (const float*)d_in[7];
  float* out = (float*)d_out;

  char* ws = (char*)d_ws;
  __bf16* Xg    = (__bf16*)(ws + XG_OFF);
  __bf16* Hbuf  = (__bf16*)(ws + HB_OFF);
  __bf16* W1b   = (__bf16*)(ws + W1B_OFF);
  __bf16* W2b   = (__bf16*)(ws + W2B_OFF);
  int*    eid   = (int*)(ws + EID_OFF);
  float*  psel  = (float*)(ws + PSEL_OFF);
  int*    rmap  = (int*)(ws + RMAP_OFF);
  int*    meta  = (int*)(ws + META_OFF);       // counts[0..3], cursors[4..7], probsum[8..11]
  int*    counts  = meta;
  int*    cursors = meta + 4;
  float*  probsum = (float*)(meta + 8);
  int*    padoff  = (int*)(ws + PADOFF_OFF);
  int2*   tiles   = (int2*)(ws + TILES_OFF);

  float* out_tail = out + (size_t)T_TOKENS * DMODEL;  // balance_loss, num_tokens[4]

  int nx4 = (int)(XG_BYTES / 16);
  clear_kernel<<<2048, 256, 0, stream>>>((uint4*)Xg, nx4, rmap, meta);
  gate_kernel<<<T_TOKENS / 8, 256, 0, stream>>>(x, gate_w, eid, psel, counts, probsum);
  setup_kernel<<<1, 32, 0, stream>>>(counts, probsum, padoff, tiles, out_tail);
  scatter_kernel<<<T_TOKENS / 8, 256, 0, stream>>>(x, eid, cursors, padoff, rmap, Xg);
  convw_kernel<<<1024, 256, 0, stream>>>(w1, w2, W1b, W2b);

  dim3 ggrid(NTILES_N, MAXMT);
  gemm_kernel<true><<<ggrid, 256, 0, stream>>>(Xg, W1b, b1, tiles, rmap, psel, Hbuf, out);
  gemm_kernel<false><<<ggrid, 256, 0, stream>>>(Hbuf, W2b, b2, tiles, rmap, psel, Hbuf, out);
}
